// SoftGatedSegmentedModel_61933428408556
// MI455X (gfx1250) — compile-verified, matrix operand order checked by reading, not run on hardware
//
#include <hip/hip_runtime.h>

// ---------------------------------------------------------------------------
// Types for CDNA5 WMMA (wave32): bf16 A/B fragments (16 x bf16 = 8 VGPRs),
// f32 accumulator (8 VGPRs).
// ---------------------------------------------------------------------------
typedef __attribute__((ext_vector_type(16))) __bf16 v16bf;
typedef __attribute__((ext_vector_type(8)))  __bf16 v8bf;
typedef __attribute__((ext_vector_type(8)))  float  v8f;

// ---------------------------------------------------------------------------
// B-matrix pre-pack: rearrange row-major W[K,N] (f32) into the CDNA5 WMMA
// B-fragment layout (ISA 7.12.2), bf16, ordered so each wave's k-walk is
// contiguous with a compile-time stride:
//   Packed index: (((nt*nKt + kt)*32 + lane)*16 + e)
//   For element e: v = e>>1, p = e&1
//     k_local = 2*(v&3) + 16*(v>>2) + 8*(lane>>4) + p
//     col     = nt*16 + (lane&15)
// Per k-step each lane reads 16 contiguous bf16 (2 x global_load_b128) and the
// kt->kt+1 stride is a constant 1024 B (immediate offsets, clause-able).
// ---------------------------------------------------------------------------
__global__ void pack_b_bf16_kernel(const float* __restrict__ W,
                                   __bf16* __restrict__ Bp, int K, int Ncols) {
  int i = blockIdx.x * blockDim.x + threadIdx.x;
  if (i >= K * Ncols) return;
  int e    = i & 15;
  int lane = (i >> 4) & 31;
  int rest = i >> 9;
  int nKt  = K >> 5;
  int kt   = rest % nKt;
  int nt   = rest / nKt;
  int v = e >> 1, p = e & 1;
  int kl  = ((v & 3) << 1) + ((v >> 2) << 4) + ((lane >> 4) << 3) + p;
  int col = (nt << 4) + (lane & 15);
  int k   = (kt << 5) + kl;
  Bp[i] = (__bf16)W[k * Ncols + col];
}

// ---------------------------------------------------------------------------
// Tiled bf16 GEMM: C[M,Ncols] = A[M,K(row-major)] x Bpacked, f32 accumulate.
// Block = 128 threads = 4 waves; wave w -> tile (blockIdx.x, 4*blockIdx.y+w).
// Requires M % 16 == 0, K % 32 == 0, Ncols % 16 == 0 (true at all call sites).
// KC > 0: compile-time K (fully unrolled); KC == 0: runtime K.
// ---------------------------------------------------------------------------
template <int KC>
__device__ inline void gemm_body(const __bf16* __restrict__ A,
                                 const __bf16* __restrict__ Bp,
                                 float* __restrict__ C,
                                 int Ncols, int Krt) {
  const int K    = (KC > 0) ? KC : Krt;
  const int nKt  = K >> 5;
  const int lane = threadIdx.x & 31;
  const int wave = threadIdx.x >> 5;
  const int m0   = blockIdx.x * 16;
  const int nTn  = Ncols >> 4;
  const int nt   = blockIdx.y * 4 + wave;
  if (nt >= nTn) return;  // wave-uniform

  // A: lane reads row (m0 + lane&15), 8 bf16 at k-offset kb and kb+16
  const __bf16* ap = A + (m0 + (lane & 15)) * K + ((lane >> 4) << 3);
  // B: contiguous per-wave k-walk, constant 512-element (1024 B) stride
  const __bf16* bp = Bp + ((nt * nKt) << 9) + (lane << 4);

  v8f acc = {};
#pragma unroll
  for (int kt = 0; kt < K / 32; ++kt) {
    const __bf16* pa = ap + (kt << 5);
    v8bf alo = *(const v8bf*)(pa);
    v8bf ahi = *(const v8bf*)(pa + 16);
    v16bf a = __builtin_shufflevector(alo, ahi, 0, 1, 2, 3, 4, 5, 6, 7,
                                      8, 9, 10, 11, 12, 13, 14, 15);
    v16bf b = *(const v16bf*)(bp + (kt << 9));
    acc = __builtin_amdgcn_wmma_f32_16x16x32_bf16(
        false, a, false, b, (short)0, acc, false, false);
  }

  // C/D layout: VGPR v -> row = m0 + v + 8*(lane>>4), col = nt*16 + (lane&15)
  float* cp = C + (m0 + ((lane >> 4) << 3)) * Ncols + (nt << 4) + (lane & 15);
#pragma unroll
  for (int v = 0; v < 8; ++v) cp[v * Ncols] = acc[v];
}

template <int KC>
__global__ void __launch_bounds__(128)
gemm_bf16_wmma_t(const __bf16* __restrict__ A, const __bf16* __restrict__ Bp,
                 float* __restrict__ C, int Ncols) {
  gemm_body<KC>(A, Bp, C, Ncols, KC);
}

__global__ void __launch_bounds__(128)
gemm_bf16_wmma_rt(const __bf16* __restrict__ A, const __bf16* __restrict__ Bp,
                  float* __restrict__ C, int K, int Ncols) {
  gemm_body<0>(A, Bp, C, Ncols, K);
}

static inline void launch_gemm(const __bf16* A, const __bf16* Bp, float* C,
                               int M, int K, int Ncols, hipStream_t stream) {
  dim3 g((M + 15) / 16, (Ncols + 63) / 64);
  if (K == 128)      gemm_bf16_wmma_t<128><<<g, 128, 0, stream>>>(A, Bp, C, Ncols);
  else if (K == 64)  gemm_bf16_wmma_t<64><<<g, 128, 0, stream>>>(A, Bp, C, Ncols);
  else if (K == 256) gemm_bf16_wmma_t<256><<<g, 128, 0, stream>>>(A, Bp, C, Ncols);
  else               gemm_bf16_wmma_rt<<<g, 128, 0, stream>>>(A, Bp, C, K, Ncols);
}

// ---------------------------------------------------------------------------
// Elementwise helpers
// ---------------------------------------------------------------------------
__global__ void f32_to_bf16_kernel(const float* __restrict__ in,
                                   __bf16* __restrict__ out, int n) {
  int i = blockIdx.x * blockDim.x + threadIdx.x;
  if (i < n) out[i] = (__bf16)in[i];
}

__global__ void zero_f32_kernel(float* __restrict__ p, int n) {
  int i = blockIdx.x * blockDim.x + threadIdx.x;
  if (i < n) p[i] = 0.0f;
}

__global__ void init_softmax_kernel(float* __restrict__ m, float* __restrict__ den, int n) {
  int i = blockIdx.x * blockDim.x + threadIdx.x;
  if (i < n) {
    m[i]   = __int_as_float((int)0xFF800000u);  // -inf
    den[i] = 0.0f;
  }
}

// per-node attention coefficients: s[n,h] = <H[n,h,:], aS[h,:]>, d likewise
__global__ void node_att_kernel(const float* __restrict__ H,
                                const float* __restrict__ aS,
                                const float* __restrict__ aD,
                                int Nn, int Hh, int C,
                                float* __restrict__ sOut, float* __restrict__ dOut) {
  int n = blockIdx.x * blockDim.x + threadIdx.x;
  if (n >= Nn) return;
  for (int h = 0; h < Hh; ++h) {
    const float* hp = H + (n * Hh + h) * C;
    float ss = 0.0f, dd = 0.0f;
    for (int c = 0; c < C; ++c) {
      float v = hp[c];
      ss = fmaf(v, aS[h * C + c], ss);
      dd = fmaf(v, aD[h * C + c], dd);
    }
    sOut[n * Hh + h] = ss;
    dOut[n * Hh + h] = dd;
  }
}

__device__ inline void edge_sd(const int* __restrict__ ei, int E, int e,
                               int& s, int& d) {
  if (e < E) { s = ei[e]; d = ei[E + e]; }
  else       { s = e - E; d = e - E; }     // appended self-loops
}

// float atomic max via ordered int/uint encoding (native atomic_max/min)
__device__ inline void atomicMaxFloat(float* addr, float v) {
  if (v >= 0.0f) atomicMax((int*)addr, __float_as_int(v));
  else           atomicMin((unsigned int*)addr, __float_as_uint(v));
}

// pass A: e = leaky_relu(s[src]+d[dst]); store, atomic-max into m[dst]
__global__ void edge_logits_max_kernel(const float* __restrict__ s_,
                                       const float* __restrict__ d_,
                                       const int* __restrict__ ei, int E, int Et,
                                       int Hh, float* __restrict__ ebuf,
                                       float* __restrict__ m) {
  int i = blockIdx.x * blockDim.x + threadIdx.x;
  if (i >= Et * Hh) return;
  int e = i / Hh, h = i - e * Hh;
  int s, d; edge_sd(ei, E, e, s, d);
  float v = s_[s * Hh + h] + d_[d * Hh + h];
  v = (v > 0.0f) ? v : 0.2f * v;  // leaky_relu, slope 0.2
  ebuf[i] = v;
  atomicMaxFloat(&m[d * Hh + h], v);
}

// pass B: ex = exp(e - m[dst]); overwrite ebuf, atomic-add denom[dst]
__global__ void edge_exp_sum_kernel(float* __restrict__ ebuf,
                                    const int* __restrict__ ei, int E, int Et,
                                    int Hh, const float* __restrict__ m,
                                    float* __restrict__ den) {
  int i = blockIdx.x * blockDim.x + threadIdx.x;
  if (i >= Et * Hh) return;
  int e = i / Hh, h = i - e * Hh;
  int s, d; edge_sd(ei, E, e, s, d);
  (void)s;
  float ex = __expf(ebuf[i] - m[d * Hh + h]);
  ebuf[i] = ex;
  atomicAdd(&den[d * Hh + h], ex);
}

// pass C: agg[dst] += (ex/denom[dst]) * H[src]; block = Hh*C threads per edge
__global__ void edge_aggregate_kernel(const float* __restrict__ Hfeat,
                                      const float* __restrict__ ex,
                                      const float* __restrict__ den,
                                      const int* __restrict__ ei, int E, int Et,
                                      int Hh, int C, float* __restrict__ agg) {
  int e = blockIdx.x;
  if (e >= Et) return;
  int s, d; edge_sd(ei, E, e, s, d);
  int t = threadIdx.x;        // 0 .. Hh*C-1
  int h = t / C;
  float alpha = ex[e * Hh + h] / den[d * Hh + h];
  float msg = alpha * Hfeat[s * (Hh * C) + t];
  atomicAdd(&agg[d * (Hh * C) + t], msg);
}

// out = elu(agg + bias); optional bf16 copy for the next GEMM input
__global__ void bias_elu_kernel(const float* __restrict__ agg,
                                const float* __restrict__ bias, int Nn, int HC,
                                float* __restrict__ outF, __bf16* __restrict__ outB) {
  int i = blockIdx.x * blockDim.x + threadIdx.x;
  if (i >= Nn * HC) return;
  float v = agg[i] + bias[i % HC];
  v = (v > 0.0f) ? v : (__expf(v) - 1.0f);  // elu
  outF[i] = v;
  if (outB) outB[i] = (__bf16)v;
}

// MLP head: y = relu(h @ W1 + b1) @ W2 + b2 ; weights staged through LDS
__global__ void fc_head_kernel(const float* __restrict__ h,
                               const float* __restrict__ W1,
                               const float* __restrict__ b1,
                               const float* __restrict__ W2,
                               const float* __restrict__ b2,
                               int Nn, int C, int Fh, float* __restrict__ y) {
  extern __shared__ float sm[];
  float* W1s = sm;              // [C*Fh]
  float* b1s = W1s + C * Fh;    // [Fh]
  float* W2s = b1s + Fh;        // [Fh]
  float* b2s = W2s + Fh;        // [1]
  for (int i = threadIdx.x; i < C * Fh; i += blockDim.x) W1s[i] = W1[i];
  for (int i = threadIdx.x; i < Fh; i += blockDim.x) { b1s[i] = b1[i]; W2s[i] = W2[i]; }
  if (threadIdx.x == 0) b2s[0] = b2[0];
  __syncthreads();

  int n = blockIdx.x * blockDim.x + threadIdx.x;
  if (n >= Nn) return;
  float z[32];
  for (int j = 0; j < Fh; ++j) z[j] = b1s[j];
  const float* hp = h + n * C;
  for (int c = 0; c < C; ++c) {
    float xv = hp[c];
    for (int j = 0; j < Fh; ++j) z[j] = fmaf(xv, W1s[c * Fh + j], z[j]);
  }
  float acc = b2s[0];
  for (int j = 0; j < Fh; ++j) acc += fmaxf(z[j], 0.0f) * W2s[j];
  y[n] = acc;
}

// gate + blend: out[0:N) = (1-g)*y0 + g*y1, out[N:2N) = g
__global__ void combine_kernel(const float* __restrict__ y, int Nn,
                               float* __restrict__ out) {
  int n = blockIdx.x * blockDim.x + threadIdx.x;
  if (n >= Nn) return;
  float g = 1.0f / (1.0f + __expf(-y[2 * Nn + n]));
  out[n] = (1.0f - g) * y[n] + g * y[Nn + n];
  out[Nn + n] = g;
}

// ---------------------------------------------------------------------------
// Host orchestration
// ---------------------------------------------------------------------------
extern "C" void kernel_launch(void* const* d_in, const int* in_sizes, int n_in,
                              void* d_out, int out_size, void* d_ws, size_t ws_size,
                              hipStream_t stream) {
  (void)n_in; (void)out_size; (void)ws_size;
  const float* x   = (const float*)d_in[0];
  const int*   ei  = (const int*)d_in[1];
  const float* c1W = (const float*)d_in[2];
  const float* c1b = (const float*)d_in[3];
  const float* a1s = (const float*)d_in[4];
  const float* a1d = (const float*)d_in[5];
  const float* c2W = (const float*)d_in[6];
  const float* c2b = (const float*)d_in[7];
  const float* a2s = (const float*)d_in[8];
  const float* a2d = (const float*)d_in[9];
  const float* f1W = (const float*)d_in[10];
  const float* f1b = (const float*)d_in[11];
  const float* f2W = (const float*)d_in[12];
  const float* f2b = (const float*)d_in[13];

  const int NB  = 3;
  const int HC1 = in_sizes[3] / NB;            // 128 (= heads*hid)
  const int HID = in_sizes[7] / NB;            // 64
  const int Hh  = HC1 / HID;                   // 2 heads (conv1)
  const int IN  = in_sizes[2] / (NB * HC1);    // 128
  const int Nn  = in_sizes[0] / IN;            // 100000
  const int E   = in_sizes[1] / 2;             // 800000
  const int Et  = E + Nn;                      // + self loops
  const int F1  = in_sizes[11] / NB;           // 32

  char* wsb = (char*)d_ws;
  size_t off = 0;
  auto alloc = [&](size_t bytes) -> void* {
    void* p = wsb + off;
    off = (off + bytes + 255) & ~(size_t)255;
    return p;
  };
  __bf16* xb  = (__bf16*)alloc((size_t)Nn * IN * 2);
  __bf16* w1b = (__bf16*)alloc((size_t)IN * HC1 * 2);   // packed fragments
  __bf16* w2b = (__bf16*)alloc((size_t)HC1 * HID * 2);  // packed fragments
  float*  H1  = (float*)alloc((size_t)Nn * HC1 * 4);    // conv1 h; reused for conv2 h
  float*  AG  = (float*)alloc((size_t)Nn * HC1 * 4);    // agg / elu outputs (reused)
  __bf16* h1b = (__bf16*)alloc((size_t)Nn * HC1 * 2);   // elu(conv1) in bf16
  float*  s1  = (float*)alloc((size_t)Nn * Hh * 4);
  float*  d1  = (float*)alloc((size_t)Nn * Hh * 4);
  float*  m1  = (float*)alloc((size_t)Nn * Hh * 4);
  float*  dn1 = (float*)alloc((size_t)Nn * Hh * 4);
  float*  eb1 = (float*)alloc((size_t)Et * Hh * 4);
  float*  s2  = (float*)alloc((size_t)Nn * 4);
  float*  d2  = (float*)alloc((size_t)Nn * 4);
  float*  m2  = (float*)alloc((size_t)Nn * 4);
  float*  dn2 = (float*)alloc((size_t)Nn * 4);
  float*  eb2 = (float*)alloc((size_t)Et * 4);
  float*  y   = (float*)alloc((size_t)NB * Nn * 4);

  const int T = 256;
  auto cdiv = [](int a, int b) { return (a + b - 1) / b; };

  f32_to_bf16_kernel<<<cdiv(Nn * IN, T), T, 0, stream>>>(x, xb, Nn * IN);

  for (int b = 0; b < NB; ++b) {
    pack_b_bf16_kernel<<<cdiv(IN * HC1, T), T, 0, stream>>>(
        c1W + (size_t)b * IN * HC1, w1b, IN, HC1);
    pack_b_bf16_kernel<<<cdiv(HC1 * HID, T), T, 0, stream>>>(
        c2W + (size_t)b * HC1 * HID, w2b, HC1, HID);

    // ---- GAT conv1: h1 = x @ W1 ----
    launch_gemm(xb, w1b, H1, Nn, IN, HC1, stream);
    node_att_kernel<<<cdiv(Nn, T), T, 0, stream>>>(
        H1, a1s + (size_t)b * Hh * HID, a1d + (size_t)b * Hh * HID,
        Nn, Hh, HID, s1, d1);
    init_softmax_kernel<<<cdiv(Nn * Hh, T), T, 0, stream>>>(m1, dn1, Nn * Hh);
    zero_f32_kernel<<<cdiv(Nn * HC1, T), T, 0, stream>>>(AG, Nn * HC1);
    edge_logits_max_kernel<<<cdiv(Et * Hh, T), T, 0, stream>>>(
        s1, d1, ei, E, Et, Hh, eb1, m1);
    edge_exp_sum_kernel<<<cdiv(Et * Hh, T), T, 0, stream>>>(
        eb1, ei, E, Et, Hh, m1, dn1);
    edge_aggregate_kernel<<<Et, Hh * HID, 0, stream>>>(
        H1, eb1, dn1, ei, E, Et, Hh, HID, AG);
    bias_elu_kernel<<<cdiv(Nn * HC1, T), T, 0, stream>>>(
        AG, c1b + (size_t)b * HC1, Nn, HC1, AG, h1b);

    // ---- GAT conv2: h2 = elu(h1) @ W2 ----
    launch_gemm(h1b, w2b, H1, Nn, HC1, HID, stream);
    node_att_kernel<<<cdiv(Nn, T), T, 0, stream>>>(
        H1, a2s + (size_t)b * HID, a2d + (size_t)b * HID, Nn, 1, HID, s2, d2);
    init_softmax_kernel<<<cdiv(Nn, T), T, 0, stream>>>(m2, dn2, Nn);
    zero_f32_kernel<<<cdiv(Nn * HID, T), T, 0, stream>>>(AG, Nn * HID);
    edge_logits_max_kernel<<<cdiv(Et, T), T, 0, stream>>>(
        s2, d2, ei, E, Et, 1, eb2, m2);
    edge_exp_sum_kernel<<<cdiv(Et, T), T, 0, stream>>>(
        eb2, ei, E, Et, 1, m2, dn2);
    edge_aggregate_kernel<<<Et, HID, 0, stream>>>(
        H1, eb2, dn2, ei, E, Et, 1, HID, AG);
    bias_elu_kernel<<<cdiv(Nn * HID, T), T, 0, stream>>>(
        AG, c2b + (size_t)b * HID, Nn, HID, AG, nullptr);

    // ---- MLP head ----
    size_t sh = (size_t)(HID * F1 + 2 * F1 + 1) * sizeof(float);
    fc_head_kernel<<<cdiv(Nn, T), T, sh, stream>>>(
        AG, f1W + (size_t)b * HID * F1, f1b + (size_t)b * F1,
        f2W + (size_t)b * F1, f2b + b, Nn, HID, F1, y + (size_t)b * Nn);
  }

  combine_kernel<<<cdiv(Nn, T), T, 0, stream>>>(y, Nn, (float*)d_out);
}
